// SequentialBPR_20177756357098
// MI455X (gfx1250) — compile-verified, matrix operand order checked by reading, not run on hardware
//
#include <hip/hip_runtime.h>
#include <hip/hip_bf16.h>
#include <math.h>

typedef __attribute__((ext_vector_type(16))) _Float16 v16h;
typedef __attribute__((ext_vector_type(8)))  float    v8f;

#define ITEMS 128
#define LOG2E   1.4426950408889634f
#define LN2     0.6931471805599453f
#define CLAMP2  28.853900817779268f   /* 20.0 * log2(e) */

__global__ void bpr_zero_kernel(float* ws) {
    ws[0] = 0.0f;   // sum of bpr over valid rows
    ws[1] = 0.0f;   // count of valid rows
}

__global__ __launch_bounds__(256)
void bpr_main_kernel(const float* __restrict__ inp,
                     const float* __restrict__ tgt,
                     float* __restrict__ ws) {
    __shared__ float xl[ITEMS];   // input row pre-scaled by log2(e)
    __shared__ float pln[ITEMS];  // pos ? ln2 : 0  (folds ln2 conversion into mask)
    __shared__ float ns[ITEMS];   // negative mask (1.0/0.0)
    __shared__ int   flags;       // bit0: any non-NaN, bit1: any pos, bit2: any neg
    __shared__ float partial[8];

    const int row = blockIdx.x;     // flattened (n, t)
    const int tid = threadIdx.x;

    if (tid == 0) flags = 0;
    __syncthreads();

    if (tid < ITEMS) {
        float x = inp[row * ITEMS + tid];
        float t = tgt[row * ITEMS + tid];
        bool nn  = (t == t);                 // !isnan
        bool pos = nn && (t > 0.5f);         // new_target > 0
        bool neg = nn && (t < 0.5f);         // new_target < 0
        xl[tid]  = x * LOG2E;
        pln[tid] = pos ? LN2 : 0.0f;
        ns[tid]  = neg ? 1.0f : 0.0f;
        int f = (nn ? 1 : 0) | (pos ? 2 : 0) | (neg ? 4 : 0);
        if (f) atomicOr(&flags, f);
    }
    __syncthreads();

    const int f = flags;  // uniform across block -> no divergence below

    if ((f & 6) == 6) {   // need >=1 positive AND >=1 negative for nonzero bpr
        const int wave = tid >> 5;
        const int lane = tid & 31;
        const int m    = lane & 15;
        const int hi   = lane >> 4;           // half-wave select
        const int i    = wave * 16 + m;       // this lane's i-row
        const float xil = xl[i];              // already log2e-scaled

        v16h ones;
        #pragma unroll
        for (int h = 0; h < 16; ++h) ones[h] = (_Float16)1.0f;

        v8f acc = {};

        // j swept in chunks of 32 -> one v_wmma_f32_16x16x32_f16 per chunk.
        // B = all-ones so D[m,n] += sum_k A[m,k].
        // A[m,k] = pos[j]*ln2 * log2(1 + 2^(min((x_i-x_j)*log2e, 20*log2e)))
        //        = pos[j] * softplus(min(x_i-x_j, 20))
        #pragma unroll
        for (int j0 = 0; j0 < ITEMS; j0 += 32) {
            v16h a;
            #pragma unroll
            for (int h = 0; h < 16; ++h) {
                // 16-bit A-matrix 16x32 layout: VGPR v = h/2, half p = h%2
                // lanes 0-15:  v<4 -> K = 2v+p      ; v>=4 -> K = 16 + 2(v-4)+p
                // lanes 16-31: same + 8
                const int v = h >> 1, p = h & 1;
                const int K = ((v < 4) ? (2 * v) : (2 * v + 8)) + p + 8 * hi;
                const int j = j0 + K;
                float d2 = xil - xl[j];                    // log2-domain diff
                d2 = fminf(d2, CLAMP2);                    // clamp (monotone in scaled domain)
                float e  = __builtin_amdgcn_exp2f(d2);     // raw v_exp_f32, no guards
                float l2 = __builtin_amdgcn_logf(1.0f + e);// raw v_log_f32; arg >= 1, no denorms
                a[h] = (_Float16)(l2 * pln[j]);            // one mul: mask * ln2 folded
            }
            acc = __builtin_amdgcn_wmma_f32_16x16x32_f16(
                /*neg_a=*/false, a, /*neg_b=*/false, ones,
                /*c_mod=*/(short)0, acc, /*reuse_a=*/false, /*reuse_b=*/false);
        }

        // C/D layout: lane l, VGPR v -> M = v + 8*hi; all N columns equal.
        // Apply the per-row neg[i] mask HERE (8 FMAs) instead of 64 inner-loop muls.
        float s = 0.0f;
        #pragma unroll
        for (int v = 0; v < 8; ++v)
            s += ns[wave * 16 + 8 * hi + v] * acc[v];
        s += __shfl_xor(s, 16, 32);                       // add other half -> chunk total
        if (lane == 0) partial[wave] = s;
    }
    __syncthreads();

    if (tid == 0) {
        float bpr = 0.0f;
        if ((f & 6) == 6) {
            for (int w = 0; w < 8; ++w) bpr += partial[w];
        }
        if (f & 1) {  // row has at least one non-NaN label
            atomicAdd(&ws[0], bpr);
            atomicAdd(&ws[1], 1.0f);
        }
    }
}

__global__ void bpr_final_kernel(const float* __restrict__ ws,
                                 float* __restrict__ out) {
    out[0] = ws[0] / ws[1];
}

extern "C" void kernel_launch(void* const* d_in, const int* in_sizes, int n_in,
                              void* d_out, int out_size, void* d_ws, size_t ws_size,
                              hipStream_t stream) {
    const float* inp = (const float*)d_in[0];   // [N,T,I] fp32
    const float* tgt = (const float*)d_in[1];   // [N,T,I] fp32 with NaNs
    float* out = (float*)d_out;                 // scalar fp32
    float* ws  = (float*)d_ws;                  // [sum_bpr, valid_count]

    const int rows = in_sizes[0] / ITEMS;       // N*T = 4096

    bpr_zero_kernel<<<1, 1, 0, stream>>>(ws);
    bpr_main_kernel<<<rows, 256, 0, stream>>>(inp, tgt, ws);
    bpr_final_kernel<<<1, 1, 0, stream>>>(ws, out);
}
